// SelfAttentionLayer_80023830659530
// MI455X (gfx1250) — matrix operand order repaired and checked by reading, not compiled, not run on hardware
//
#include <hip/hip_runtime.h>
#include <stdint.h>

#define N_TOK  8192
#define DMODEL 128
#define BR     128     // Q rows per workgroup (8 waves x 16 rows)
#define BC     64      // K/V rows per inner iteration

typedef __attribute__((ext_vector_type(16))) __bf16 bf16x16;
typedef __attribute__((ext_vector_type(8)))  float  f32x8;

union Frag {
  uint32_t u[8];
  __bf16   h[16];
  bf16x16  v;
};

__device__ __forceinline__ f32x8 wmma_bf16(bf16x16 a, bf16x16 b, f32x8 c) {
  // D = A(16x32 bf16) * B(32x16 bf16) + C(16x16 f32)
  return __builtin_amdgcn_wmma_f32_16x16x32_bf16(false, a, false, b, (short)0, c, false, false);
}

// A-fragment (16x32, MxK) from row-major bf16 (ld = leading dim in elements).
// 16-bit A layout (contiguous k, matches B loaders): lane L -> M = L&15; VGPR v -> K = 16*(L>=16) + 2*v.
__device__ __forceinline__ bf16x16 load_a_frag(const __bf16* src, int ld, int row0, int k0) {
  Frag f;
  int l = threadIdx.x & 31;
  const __bf16* rowp = src + (size_t)(row0 + (l & 15)) * ld;
  int khalf = (l >> 4) * 16;
#pragma unroll
  for (int v = 0; v < 8; ++v) {
    int k = k0 + khalf + v * 2;
    f.u[v] = *(const uint32_t*)(rowp + k);       // packs bf16 pair (k, k+1)
  }
  return f.v;
}

// B-fragment (32x16, KxN) where B[k][n] = src[(n0+n)*ld + k]  (i.e. B = stored^T).
// B layout used here (contiguous k, matches A loader): lane L -> N = L&15, K base = 16*(L>=16); VGPR v -> K = 2v, 2v+1.
__device__ __forceinline__ bf16x16 load_bT_frag(const __bf16* src, int ld, int n0, int k0) {
  Frag f;
  int l = threadIdx.x & 31;
  const __bf16* rowp = src + (size_t)(n0 + (l & 15)) * ld + k0 + ((l >> 4) << 4);
#pragma unroll
  for (int v = 0; v < 8; ++v) f.u[v] = *(const uint32_t*)(rowp + 2 * v);
  return f.v;
}

// B-fragment (32x16, KxN) where B[k][n] = src[(k0+k)*ld + (n0+n)] (true row-major B).
__device__ __forceinline__ bf16x16 load_b_frag(const __bf16* src, int ld, int k0, int n0) {
  Frag f;
  int l = threadIdx.x & 31;
  int n = n0 + (l & 15);
  int kb = k0 + ((l >> 4) << 4);
#pragma unroll
  for (int v = 0; v < 8; ++v) {
    f.h[2 * v]     = src[(size_t)(kb + 2 * v) * ld + n];
    f.h[2 * v + 1] = src[(size_t)(kb + 2 * v + 1) * ld + n];
  }
  return f.v;
}

// Async global -> LDS copy, 16 bytes per lane. Tracked by ASYNCcnt.
// VDST VGPR holds the LDS byte offset (LDS_BASE added by HW), GV addressing.
__device__ __forceinline__ void async_g2l_16B(const __bf16* g, uint32_t lds_byte_off) {
  asm volatile("global_load_async_to_lds_b128 %0, %1, off"
               :: "v"(lds_byte_off), "v"(g) : "memory");
}

__device__ __forceinline__ void wait_asynccnt0() {
  asm volatile("s_wait_asynccnt 0x0" ::: "memory");
}

// LDS byte offset of a shared-memory pointer: flat LDS-aperture addr[31:0].
__device__ __forceinline__ uint32_t lds_offset(const void* p) {
  return (uint32_t)(uintptr_t)p;
}

__global__ void cvt_f32_bf16_kernel(const float* __restrict__ src,
                                    __bf16* __restrict__ dst, int n) {
  int i = blockIdx.x * blockDim.x + threadIdx.x;
  if (i < n) dst[i] = (__bf16)src[i];
}

// Q = X Wq^T, K = X Wk^T, V = X Wv^T  (all [N,128] bf16 out, f32 accumulate)
__global__ __launch_bounds__(256)
void qkv_proj_kernel(const __bf16* __restrict__ Xb,
                     const __bf16* __restrict__ Wqb,
                     const __bf16* __restrict__ Wkb,
                     const __bf16* __restrict__ Wvb,
                     __bf16* __restrict__ Qb,
                     __bf16* __restrict__ Kb,
                     __bf16* __restrict__ Vb) {
  int wave = threadIdx.x >> 5;
  int l    = threadIdx.x & 31;
  int row0 = blockIdx.x * BR + wave * 16;
  int nlane = l & 15;
  int half8 = (l >> 4) * 8;

  bf16x16 afr[4];
#pragma unroll
  for (int ks = 0; ks < 4; ++ks) afr[ks] = load_a_frag(Xb, DMODEL, row0, ks * 32);

  const __bf16* Ws[3] = {Wqb, Wkb, Wvb};
  __bf16*       Os[3] = {Qb, Kb, Vb};
#pragma unroll
  for (int t = 0; t < 3; ++t) {
#pragma unroll
    for (int nt = 0; nt < 8; ++nt) {
      f32x8 acc = {};
#pragma unroll
      for (int ks = 0; ks < 4; ++ks)
        acc = wmma_bf16(afr[ks], load_bT_frag(Ws[t], DMODEL, nt * 16, ks * 32), acc);
#pragma unroll
      for (int v = 0; v < 8; ++v)
        Os[t][(size_t)(row0 + half8 + v) * DMODEL + nt * 16 + nlane] = (__bf16)acc[v];
    }
  }
}

// Fused flash attention with async double-buffered K/V staging.
// Dynamic LDS layout (80 KB):
//   [buf0: K 16KB][buf0: V 16KB][buf1: K 16KB][buf1: V 16KB][P: 8 x 2KB]
#define MATB   (BC * DMODEL * 2)        // 16384 bytes per staged matrix
#define SMEMSZ (4 * MATB + 8 * 16 * BC * 2)

__global__ __launch_bounds__(256)
void flash_attn_kernel(const __bf16* __restrict__ Qb,
                       const __bf16* __restrict__ Kb,
                       const __bf16* __restrict__ Vb,
                       float* __restrict__ Out) {
  extern __shared__ __align__(16) char smem[];

  int tid  = threadIdx.x;
  int wave = tid >> 5;
  int l    = tid & 31;
  int row0 = blockIdx.x * BR + wave * 16;
  int nlane = l & 15;
  int half8 = (l >> 4) * 8;

  __bf16* pw     = (__bf16*)(smem + 4 * MATB) + wave * 16 * BC;
  uint32_t smem0 = lds_offset(smem);

  bf16x16 qfr[4];
#pragma unroll
  for (int ks = 0; ks < 4; ++ks) qfr[ks] = load_a_frag(Qb, DMODEL, row0, ks * 32);

  f32x8 o[8];
#pragma unroll
  for (int d = 0; d < 8; ++d) o[d] = {};
  float mrow[8], lrow[8];
#pragma unroll
  for (int v = 0; v < 8; ++v) { mrow[v] = -__builtin_inff(); lrow[v] = 0.0f; }

  const int NBLK = N_TOK / BC;  // 128

  // Prologue: stage block 0 into buffer 0 (async DMA, then drain + barrier).
  {
#pragma unroll
    for (int i = 0; i < 4; ++i) {                // 1024 x 16B per matrix, 256 thr
      int t16 = tid + i * 256;
      async_g2l_16B(Kb + t16 * 8, smem0 + (uint32_t)(t16 * 16));
      async_g2l_16B(Vb + t16 * 8, smem0 + MATB + (uint32_t)(t16 * 16));
    }
    wait_asynccnt0();
    __syncthreads();
  }

  for (int jb = 0; jb < NBLK; ++jb) {
    int cur = jb & 1;
    int nxt = cur ^ 1;

    // Current buffers (computed arithmetically; no LDS pointer arrays).
    const __bf16* Kcur = (const __bf16*)(smem + (size_t)cur * 2 * MATB);
    const __bf16* Vcur = (const __bf16*)(smem + (size_t)cur * 2 * MATB + MATB);

    // Kick off async staging of the NEXT block while we compute on this one.
    if (jb + 1 < NBLK) {
      const __bf16* gK = Kb + (size_t)(jb + 1) * BC * DMODEL;
      const __bf16* gV = Vb + (size_t)(jb + 1) * BC * DMODEL;
      uint32_t kb_off = smem0 + (uint32_t)(nxt * 2 * MATB);
#pragma unroll
      for (int i = 0; i < 4; ++i) {
        int t16 = tid + i * 256;
        async_g2l_16B(gK + t16 * 8, kb_off + (uint32_t)(t16 * 16));
        async_g2l_16B(gV + t16 * 8, kb_off + MATB + (uint32_t)(t16 * 16));
      }
    }

    // S(16xBC) = Q(16x128) * K^T(128xBC), fp32 accum
    f32x8 s[4];
#pragma unroll
    for (int nt = 0; nt < 4; ++nt) {
      s[nt] = {};
#pragma unroll
      for (int ks = 0; ks < 4; ++ks)
        s[nt] = wmma_bf16(qfr[ks], load_bT_frag(Kcur, DMODEL, nt * 16, ks * 32), s[nt]);
    }

    // Online softmax per row (row v+half8 lives in one 16-lane group).
#pragma unroll
    for (int v = 0; v < 8; ++v) {
      float rm = s[0][v];
#pragma unroll
      for (int nt = 1; nt < 4; ++nt) rm = fmaxf(rm, s[nt][v]);
#pragma unroll
      for (int msk = 1; msk < 16; msk <<= 1) rm = fmaxf(rm, __shfl_xor(rm, msk, 32));
      float mn    = fmaxf(mrow[v], rm);
      float alpha = __expf(mrow[v] - mn);
      mrow[v] = mn;
      float rs = 0.0f;
#pragma unroll
      for (int nt = 0; nt < 4; ++nt) {
        float p = __expf(s[nt][v] - mn);
        s[nt][v] = p;
        rs += p;
      }
#pragma unroll
      for (int msk = 1; msk < 16; msk <<= 1) rs += __shfl_xor(rs, msk, 32);
      lrow[v] = lrow[v] * alpha + rs;
#pragma unroll
      for (int d = 0; d < 8; ++d) o[d][v] *= alpha;
    }

    // P (C-layout) -> per-wave LDS tile (row-major) -> A-layout fragments.
#pragma unroll
    for (int nt = 0; nt < 4; ++nt)
#pragma unroll
      for (int v = 0; v < 8; ++v)
        pw[(half8 + v) * BC + nt * 16 + nlane] = (__bf16)s[nt][v];

    // O(16x128) += P(16xBC) * V(BCx128)   (same-wave LDS RAW: compiler waits DScnt)
#pragma unroll
    for (int ks = 0; ks < BC / 32; ++ks) {
      bf16x16 a = load_a_frag(pw, BC, 0, ks * 32);
#pragma unroll
      for (int dt = 0; dt < 8; ++dt)
        o[dt] = wmma_bf16(a, load_b_frag(Vcur, DMODEL, ks * 32, dt * 16), o[dt]);
    }

    // Drain this wave's async copies for the next block, then block barrier so
    // every wave sees the staged data (and nobody still reads the old buffer).
    wait_asynccnt0();
    __syncthreads();
  }

  // Normalize and write fp32 output.
#pragma unroll
  for (int v = 0; v < 8; ++v) {
    float inv = 1.0f / lrow[v];
#pragma unroll
    for (int dt = 0; dt < 8; ++dt)
      Out[(size_t)(row0 + half8 + v) * DMODEL + dt * 16 + nlane] = o[dt][v] * inv;
  }
}

extern "C" void kernel_launch(void* const* d_in, const int* in_sizes, int n_in,
                              void* d_out, int out_size, void* d_ws, size_t ws_size,
                              hipStream_t stream) {
  (void)in_sizes; (void)n_in; (void)out_size; (void)ws_size;
  const float* X  = (const float*)d_in[0];
  const float* Wq = (const float*)d_in[1];
  const float* Wk = (const float*)d_in[2];
  const float* Wv = (const float*)d_in[3];
  float* Out = (float*)d_out;

  const size_t szX = (size_t)N_TOK * DMODEL;    // 1,048,576 elems
  const size_t szW = (size_t)DMODEL * DMODEL;   // 16,384 elems
  char* p = (char*)d_ws;
  __bf16* Xb  = (__bf16*)(p + 0 * 2 * szX);
  __bf16* Qb  = (__bf16*)(p + 1 * 2 * szX);
  __bf16* Kb  = (__bf16*)(p + 2 * 2 * szX);
  __bf16* Vb  = (__bf16*)(p + 3 * 2 * szX);
  __bf16* Wqb = (__bf16*)(p + 4 * 2 * szX);
  __bf16* Wkb = (__bf16*)(p + 4 * 2 * szX + 2 * szW);
  __bf16* Wvb = (__bf16*)(p + 4 * 2 * szX + 4 * szW);

  // 1) fp32 -> bf16 conversions
  cvt_f32_bf16_kernel<<<(int)((szX + 255) / 256), 256, 0, stream>>>(X,  Xb,  (int)szX);
  cvt_f32_bf16_kernel<<<(int)((szW + 255) / 256), 256, 0, stream>>>(Wq, Wqb, (int)szW);
  cvt_f32_bf16_kernel<<<(int)((szW + 255) / 256), 256, 0, stream>>>(Wk, Wkb, (int)szW);
  cvt_f32_bf16_kernel<<<(int)((szW + 255) / 256), 256, 0, stream>>>(Wv, Wvb, (int)szW);

  // 2) QKV projections (WMMA bf16)
  qkv_proj_kernel<<<N_TOK / BR, 256, 0, stream>>>(Xb, Wqb, Wkb, Wvb, Qb, Kb, Vb);

  // 3) Fused flash attention (WMMA bf16 + async double-buffered staging)
  flash_attn_kernel<<<N_TOK / BR, 256, SMEMSZ, stream>>>(Qb, Kb, Vb, Out);
}
